// SimpleSSM_13005160972578
// MI455X (gfx1250) — matrix operand order repaired and checked by reading, not compiled, not run on hardware
//
#include <hip/hip_runtime.h>

// SimpleSSM on MI455X (gfx1250, wave32, WMMA).
// Chunked parallel scan (L=16, 128 chunks) + bf16 WMMA (f32 acc).
// Critical path: 16 + 128 + 16 dependent matmul steps instead of 2048.
// Weights (512KB each in bf16) stay resident in the 192MB L2.
// Round 2: batched B-fragment loads (clause-friendly) + 2 M-tiles per block
// sharing B-fragments (8 wmma per 4 b128-load-pairs -> half the L2 weight
// traffic in the scan phases).
//
// Workspace (~104 MiB of d_ws):
//   Bu/xlocal fp32 (t,b,s)  : 64 MiB   (in-place overwritten by phase 1)
//   xs bf16 (t,b,s)         : 32 MiB
//   A,B,C bf16 + A^2..A^16  : 3.5 MiB
//   chunk states S_c fp32   : 4 MiB

constexpr int DIM    = 512;
constexpr int NB     = 16;    // batch == WMMA M
constexpr int TSEQ   = 2048;
constexpr int CHUNK  = 16;
constexpr int NCHUNK = TSEQ / CHUNK;  // 128
constexpr int LDP    = 520;   // padded LDS row stride (bf16) -> conflict-free

typedef __bf16 bf16_t;
typedef __attribute__((ext_vector_type(16))) __bf16 v16bf;
typedef __attribute__((ext_vector_type(8)))  float  v8f;

static __device__ __forceinline__ v8f wmma_bf16(v16bf a, v16bf b, v8f c) {
  return __builtin_amdgcn_wmma_f32_16x16x32_bf16(false, a, false, b, (short)0, c,
                                                 false, false);
}

// ---- WMMA operand fragment loaders (k = 16h + s in both A and B, wave32) -------------

// A-matrix 16x32 (MxK), bf16 source. lanes 0-15: M=lane; K = 16h + {0..15}.
static __device__ __forceinline__ v16bf afrag(const bf16_t* base, int ld, int lane) {
  const int m = lane & 15, h = lane >> 4;
  const bf16_t* p = base + (size_t)m * ld + 16 * h;
  v16bf a;
#pragma unroll
  for (int j = 0; j < 16; ++j) a[j] = p[j];
  return a;
}

// Same, converting from fp32 source on the fly (for u).
static __device__ __forceinline__ v16bf afrag_f32(const float* base, int ld, int lane) {
  const int m = lane & 15, h = lane >> 4;
  const float* p = base + (size_t)m * ld + 16 * h;
  v16bf a;
#pragma unroll
  for (int j = 0; j < 16; ++j) a[j] = (bf16_t)p[j];
  return a;
}

// B-matrix 32x16 (KxN) for Act @ W^T with W row-major [n][k]:
// lane n=l&15 reads 16 CONTIGUOUS bf16 from row n at k=16h (two b128 loads).
static __device__ __forceinline__ v16bf bfrag_nk(const bf16_t* Wtile, int ld, int lane) {
  const int n = lane & 15, h = lane >> 4;
  const bf16_t* p = Wtile + (size_t)n * ld + 16 * h;
  v16bf b;
#pragma unroll
  for (int j = 0; j < 16; ++j) b[j] = p[j];
  return b;
}

// B-matrix 32x16 (KxN) with W row-major [k][n] (Act @ W). Strided; only used
// by the 4 tiny A^16 squaring GEMMs.
static __device__ __forceinline__ v16bf bfrag_kn(const bf16_t* Wtile, int ld, int lane) {
  const int n = lane & 15, h = lane >> 4;
  const bf16_t* p = Wtile + (size_t)(16 * h) * ld + n;
  v16bf b;
#pragma unroll
  for (int j = 0; j < 16; ++j) b[j] = p[(size_t)j * ld];
  return b;
}

// ---- kernels ---------------------------------------------------------------

__global__ __launch_bounds__(256) void k_cvt(const float* __restrict__ A,
                                             const float* __restrict__ B,
                                             const float* __restrict__ C,
                                             bf16_t* __restrict__ Ab,
                                             bf16_t* __restrict__ Bb,
                                             bf16_t* __restrict__ Cb) {
  int i = blockIdx.x * 256 + threadIdx.x;
  if (i < DIM * DIM) {
    Ab[i] = (bf16_t)A[i];
    Bb[i] = (bf16_t)B[i];
    Cb[i] = (bf16_t)C[i];
  }
}

// Bu[t][b][s] = sum_i u[b][t][i] * B[s][i].  One block = 32 consecutive t
// (two 16-row M-tiles sharing B-fragments) for one batch; 8 waves cover s.
__global__ __launch_bounds__(256) void k_proj_in(const float* __restrict__ u,
                                                 const bf16_t* __restrict__ Wb,
                                                 float* __restrict__ Bu) {
  const int lane = threadIdx.x & 31, wave = threadIdx.x >> 5;
  const int n = lane & 15, h = lane >> 4, nbase = wave * 64;
  const int batch = blockIdx.x / (TSEQ / 32);
  const int tbase = (blockIdx.x % (TSEQ / 32)) * 32;
  const float* act0 = u + ((size_t)batch * TSEQ + tbase) * DIM;
  const float* act1 = act0 + (size_t)16 * DIM;
  v8f acc0[4] = {}, acc1[4] = {};
  for (int kb = 0; kb < DIM; kb += 32) {
    v16bf a0 = afrag_f32(act0 + kb, DIM, lane);
    v16bf a1 = afrag_f32(act1 + kb, DIM, lane);
    v16bf b[4];
#pragma unroll
    for (int nt = 0; nt < 4; ++nt)
      b[nt] = bfrag_nk(Wb + (size_t)(nbase + nt * 16) * DIM + kb, DIM, lane);
#pragma unroll
    for (int nt = 0; nt < 4; ++nt) acc0[nt] = wmma_bf16(a0, b[nt], acc0[nt]);
#pragma unroll
    for (int nt = 0; nt < 4; ++nt) acc1[nt] = wmma_bf16(a1, b[nt], acc1[nt]);
  }
#pragma unroll
  for (int nt = 0; nt < 4; ++nt)
#pragma unroll
    for (int r = 0; r < 8; ++r) {
      const int s = nbase + nt * 16 + n;
      const int t0 = tbase + r + 8 * h;
      Bu[((size_t)t0 * NB + batch) * DIM + s] = acc0[nt][r];
      Bu[((size_t)(t0 + 16) * NB + batch) * DIM + s] = acc1[nt][r];
    }
}

// Dst = X @ W, all row-major bf16 (512x512x512). Used 4x for A^16.
__global__ __launch_bounds__(256) void k_matsq(const bf16_t* __restrict__ X,
                                               const bf16_t* __restrict__ W,
                                               bf16_t* __restrict__ Dst) {
  const int lane = threadIdx.x & 31, wave = threadIdx.x >> 5;
  const int n = lane & 15, h = lane >> 4, nbase = wave * 64;
  const int rowbase = blockIdx.x * 16;
  v8f acc[4] = {};
  for (int kb = 0; kb < DIM; kb += 32) {
    v16bf a = afrag(X + (size_t)rowbase * DIM + kb, DIM, lane);
    v16bf b[4];
#pragma unroll
    for (int nt = 0; nt < 4; ++nt)
      b[nt] = bfrag_kn(W + (size_t)kb * DIM + nbase + nt * 16, DIM, lane);
#pragma unroll
    for (int nt = 0; nt < 4; ++nt) acc[nt] = wmma_bf16(a, b[nt], acc[nt]);
  }
#pragma unroll
  for (int nt = 0; nt < 4; ++nt)
#pragma unroll
    for (int r = 0; r < 8; ++r)
      Dst[(size_t)(rowbase + r + 8 * h) * DIM + nbase + nt * 16 + n] = (bf16_t)acc[nt][r];
}

// Phase 1: per-chunk local scan (zero init), in place over Bu.
// Two chunks per block share the A-matrix B-fragments.
__global__ __launch_bounds__(256) void k_scan_local(float* __restrict__ Bu,
                                                    const bf16_t* __restrict__ Ab) {
  __shared__ bf16_t Xs[2][2][16 * LDP];   // [buffer][chunk-slot][16 x 512 state]
  const int lane = threadIdx.x & 31, wave = threadIdx.x >> 5;
  const int n = lane & 15, h = lane >> 4, nbase = wave * 64;
  const int c0 = blockIdx.x * 2;
  for (int i = threadIdx.x; i < 16 * LDP; i += 256) {
    Xs[0][0][i] = (bf16_t)0.f;
    Xs[0][1][i] = (bf16_t)0.f;
  }
  __syncthreads();
  int cur = 0;
  for (int k = 0; k < CHUNK; ++k) {
    float* Bu0 = Bu + ((size_t)c0 * CHUNK + k) * (NB * DIM);
    float* Bu1 = Bu0 + (size_t)CHUNK * (NB * DIM);
    v8f acc0[4], acc1[4];
#pragma unroll
    for (int nt = 0; nt < 4; ++nt)
#pragma unroll
      for (int r = 0; r < 8; ++r) {
        const size_t idx = (size_t)(r + 8 * h) * DIM + nbase + nt * 16 + n;
        acc0[nt][r] = Bu0[idx];
        acc1[nt][r] = Bu1[idx];
      }
    for (int kb = 0; kb < DIM; kb += 32) {
      v16bf a0 = afrag(&Xs[cur][0][kb], LDP, lane);
      v16bf a1 = afrag(&Xs[cur][1][kb], LDP, lane);
      v16bf b[4];
#pragma unroll
      for (int nt = 0; nt < 4; ++nt)
        b[nt] = bfrag_nk(Ab + (size_t)(nbase + nt * 16) * DIM + kb, DIM, lane);
#pragma unroll
      for (int nt = 0; nt < 4; ++nt) acc0[nt] = wmma_bf16(a0, b[nt], acc0[nt]);
#pragma unroll
      for (int nt = 0; nt < 4; ++nt) acc1[nt] = wmma_bf16(a1, b[nt], acc1[nt]);
    }
#pragma unroll
    for (int nt = 0; nt < 4; ++nt)
#pragma unroll
      for (int r = 0; r < 8; ++r) {
        const int row = r + 8 * h, col = nbase + nt * 16 + n;
        const float v0 = acc0[nt][r], v1 = acc1[nt][r];
        Bu0[(size_t)row * DIM + col] = v0;            // xlocal (fp32, in place)
        Bu1[(size_t)row * DIM + col] = v1;
        Xs[cur ^ 1][0][row * LDP + col] = (bf16_t)v0; // next-step states
        Xs[cur ^ 1][1][row * LDP + col] = (bf16_t)v1;
      }
    __syncthreads();
    cur ^= 1;
  }
}

// Phase 2: inter-chunk recurrence S_c = E_c + S_{c-1} @ (A^16)^T.
// Single block, 128 serial steps (the critical path).
__global__ __launch_bounds__(256) void k_scan_chunks(const float* __restrict__ xlocal,
                                                     const bf16_t* __restrict__ P,
                                                     float* __restrict__ Sarr) {
  __shared__ bf16_t Xs[2][16 * LDP];
  const int lane = threadIdx.x & 31, wave = threadIdx.x >> 5;
  const int n = lane & 15, h = lane >> 4, nbase = wave * 64;
  for (int i = threadIdx.x; i < 16 * LDP; i += 256) Xs[0][i] = (bf16_t)0.f;
  __syncthreads();
  int cur = 0;
  for (int c = 0; c < NCHUNK; ++c) {
    const float* Ec = xlocal + ((size_t)c * CHUNK + (CHUNK - 1)) * (NB * DIM);
    v8f acc[4];
#pragma unroll
    for (int nt = 0; nt < 4; ++nt)
#pragma unroll
      for (int r = 0; r < 8; ++r)
        acc[nt][r] = Ec[(size_t)(r + 8 * h) * DIM + nbase + nt * 16 + n];
    for (int kb = 0; kb < DIM; kb += 32) {
      v16bf a = afrag(&Xs[cur][kb], LDP, lane);
      v16bf b[4];
#pragma unroll
      for (int nt = 0; nt < 4; ++nt)
        b[nt] = bfrag_nk(P + (size_t)(nbase + nt * 16) * DIM + kb, DIM, lane);
#pragma unroll
      for (int nt = 0; nt < 4; ++nt) acc[nt] = wmma_bf16(a, b[nt], acc[nt]);
    }
#pragma unroll
    for (int nt = 0; nt < 4; ++nt)
#pragma unroll
      for (int r = 0; r < 8; ++r) {
        const float v = acc[nt][r];
        const int row = r + 8 * h, col = nbase + nt * 16 + n;
        Sarr[(size_t)c * (NB * DIM) + (size_t)row * DIM + col] = v;
        Xs[cur ^ 1][row * LDP + col] = (bf16_t)v;
      }
    __syncthreads();
    cur ^= 1;
  }
}

// Phase 3: carry propagation + combine; two chunks per block share B-frags.
// carry_k = carry_{k-1} @ A^T from S_{chunk-1}; x_t = xlocal_t + carry.
__global__ __launch_bounds__(256) void k_scan_fix(const float* __restrict__ xlocal,
                                                  const bf16_t* __restrict__ Ab,
                                                  const float* __restrict__ Sarr,
                                                  bf16_t* __restrict__ xs_out) {
  __shared__ bf16_t Xs[2][2][16 * LDP];
  const int lane = threadIdx.x & 31, wave = threadIdx.x >> 5;
  const int n = lane & 15, h = lane >> 4, nbase = wave * 64;
  const int c0 = blockIdx.x * 2;
  // init carries = S_{c-1} (chunk 0: zeros)
  if (c0 == 0) {
    for (int i = threadIdx.x; i < 16 * LDP; i += 256) Xs[0][0][i] = (bf16_t)0.f;
  } else {
    const float* S = Sarr + (size_t)(c0 - 1) * (NB * DIM);
    for (int i = threadIdx.x; i < NB * DIM; i += 256)
      Xs[0][0][(i / DIM) * LDP + (i % DIM)] = (bf16_t)S[i];
  }
  {
    const float* S = Sarr + (size_t)c0 * (NB * DIM);
    for (int i = threadIdx.x; i < NB * DIM; i += 256)
      Xs[0][1][(i / DIM) * LDP + (i % DIM)] = (bf16_t)S[i];
  }
  __syncthreads();
  int cur = 0;
  for (int k = 0; k < CHUNK; ++k) {
    const size_t t0 = (size_t)c0 * CHUNK + k;
    const float* XL0 = xlocal + t0 * (NB * DIM);
    const float* XL1 = XL0 + (size_t)CHUNK * (NB * DIM);
    bf16_t* XO0 = xs_out + t0 * (NB * DIM);
    bf16_t* XO1 = XO0 + (size_t)CHUNK * (NB * DIM);
    v8f acc0[4] = {}, acc1[4] = {};
    for (int kb = 0; kb < DIM; kb += 32) {
      v16bf a0 = afrag(&Xs[cur][0][kb], LDP, lane);
      v16bf a1 = afrag(&Xs[cur][1][kb], LDP, lane);
      v16bf b[4];
#pragma unroll
      for (int nt = 0; nt < 4; ++nt)
        b[nt] = bfrag_nk(Ab + (size_t)(nbase + nt * 16) * DIM + kb, DIM, lane);
#pragma unroll
      for (int nt = 0; nt < 4; ++nt) acc0[nt] = wmma_bf16(a0, b[nt], acc0[nt]);
#pragma unroll
      for (int nt = 0; nt < 4; ++nt) acc1[nt] = wmma_bf16(a1, b[nt], acc1[nt]);
    }
#pragma unroll
    for (int nt = 0; nt < 4; ++nt)
#pragma unroll
      for (int r = 0; r < 8; ++r) {
        const int row = r + 8 * h, col = nbase + nt * 16 + n;
        const size_t idx = (size_t)row * DIM + col;
        const float cr0 = acc0[nt][r], cr1 = acc1[nt][r];
        XO0[idx] = (bf16_t)(cr0 + XL0[idx]);
        XO1[idx] = (bf16_t)(cr1 + XL1[idx]);
        Xs[cur ^ 1][0][row * LDP + col] = (bf16_t)cr0;
        Xs[cur ^ 1][1][row * LDP + col] = (bf16_t)cr1;
      }
    __syncthreads();
    cur ^= 1;
  }
}

// y[b][t][o] = sum_s xs[t][b][s] * C[o][s].  Two t per block share C-frags.
__global__ __launch_bounds__(256) void k_proj_out(const bf16_t* __restrict__ xs,
                                                  const bf16_t* __restrict__ Wc,
                                                  float* __restrict__ y) {
  const int lane = threadIdx.x & 31, wave = threadIdx.x >> 5;
  const int n = lane & 15, h = lane >> 4, nbase = wave * 64;
  const int t0 = blockIdx.x * 2;
  const bf16_t* act0 = xs + (size_t)t0 * NB * DIM;
  const bf16_t* act1 = act0 + (size_t)NB * DIM;
  v8f acc0[4] = {}, acc1[4] = {};
  for (int kb = 0; kb < DIM; kb += 32) {
    v16bf a0 = afrag(act0 + kb, DIM, lane);
    v16bf a1 = afrag(act1 + kb, DIM, lane);
    v16bf b[4];
#pragma unroll
    for (int nt = 0; nt < 4; ++nt)
      b[nt] = bfrag_nk(Wc + (size_t)(nbase + nt * 16) * DIM + kb, DIM, lane);
#pragma unroll
    for (int nt = 0; nt < 4; ++nt) acc0[nt] = wmma_bf16(a0, b[nt], acc0[nt]);
#pragma unroll
    for (int nt = 0; nt < 4; ++nt) acc1[nt] = wmma_bf16(a1, b[nt], acc1[nt]);
  }
#pragma unroll
  for (int nt = 0; nt < 4; ++nt)
#pragma unroll
    for (int r = 0; r < 8; ++r) {
      const int bb = r + 8 * h;
      const int o = nbase + nt * 16 + n;
      y[((size_t)bb * TSEQ + t0) * DIM + o] = acc0[nt][r];
      y[((size_t)bb * TSEQ + t0 + 1) * DIM + o] = acc1[nt][r];
    }
}

// ---- launcher --------------------------------------------------------------

extern "C" void kernel_launch(void* const* d_in, const int* in_sizes, int n_in,
                              void* d_out, int out_size, void* d_ws, size_t ws_size,
                              hipStream_t stream) {
  (void)in_sizes; (void)n_in; (void)out_size; (void)ws_size;
  const float* u = (const float*)d_in[0];
  const float* A = (const float*)d_in[1];
  const float* B = (const float*)d_in[2];
  const float* C = (const float*)d_in[3];
  float* y = (float*)d_out;

  char* ws = (char*)d_ws;
  constexpr size_t SZ_BU = (size_t)TSEQ * NB * DIM * 4;   // 64 MiB fp32
  constexpr size_t SZ_XS = (size_t)TSEQ * NB * DIM * 2;   // 32 MiB bf16
  constexpr size_t SZ_W  = (size_t)DIM * DIM * 2;         // 512 KiB bf16
  float*  Bu   = (float*)(ws);
  bf16_t* xs   = (bf16_t*)(ws + SZ_BU);
  bf16_t* Ab   = (bf16_t*)(ws + SZ_BU + SZ_XS);
  bf16_t* Bb   = (bf16_t*)(ws + SZ_BU + SZ_XS + 1 * SZ_W);
  bf16_t* Cb   = (bf16_t*)(ws + SZ_BU + SZ_XS + 2 * SZ_W);
  bf16_t* P2   = (bf16_t*)(ws + SZ_BU + SZ_XS + 3 * SZ_W);
  bf16_t* P4   = (bf16_t*)(ws + SZ_BU + SZ_XS + 4 * SZ_W);
  bf16_t* P8   = (bf16_t*)(ws + SZ_BU + SZ_XS + 5 * SZ_W);
  bf16_t* P16  = (bf16_t*)(ws + SZ_BU + SZ_XS + 6 * SZ_W);
  float*  Sarr = (float*)(ws + SZ_BU + SZ_XS + 7 * SZ_W); // 4 MiB fp32

  k_cvt<<<(DIM * DIM + 255) / 256, 256, 0, stream>>>(A, B, C, Ab, Bb, Cb);
  k_proj_in<<<NB * (TSEQ / 32), 256, 0, stream>>>(u, Bb, Bu);
  k_matsq<<<DIM / 16, 256, 0, stream>>>(Ab, Ab, P2);
  k_matsq<<<DIM / 16, 256, 0, stream>>>(P2, P2, P4);
  k_matsq<<<DIM / 16, 256, 0, stream>>>(P4, P4, P8);
  k_matsq<<<DIM / 16, 256, 0, stream>>>(P8, P8, P16);
  k_scan_local<<<NCHUNK / 2, 256, 0, stream>>>(Bu, Ab);
  k_scan_chunks<<<1, 256, 0, stream>>>(Bu, P16, Sarr);
  k_scan_fix<<<NCHUNK / 2, 256, 0, stream>>>(Bu, Ab, Sarr, xs);
  k_proj_out<<<TSEQ / 2, 256, 0, stream>>>(xs, Cb, y);
}
